// UnifiedTransducerBlock_18511309046446
// MI455X (gfx1250) — compile-verified, matrix operand order checked by reading, not run on hardware
//
#include <hip/hip_runtime.h>
#include <stdint.h>
#include <stddef.h>

// ---------------------------------------------------------------------------
// Types for CDNA5 WMMA (wave32): v_wmma_f32_16x16x32_bf16
// ---------------------------------------------------------------------------
typedef __bf16 bf16_t;
typedef __attribute__((ext_vector_type(16))) __bf16 v16bf;
typedef __attribute__((ext_vector_type(8)))  __bf16 bf16x8;
typedef __attribute__((ext_vector_type(8)))  float  v8f;
typedef __attribute__((ext_vector_type(4)))  uint32_t u32x4;
typedef __attribute__((ext_vector_type(8)))  uint32_t u32x8;

#define DMODEL 1024
#define FOURD  4096
#define FDIM   4096
#define NHEAD  16
#define DHEAD  64
#define LSEQ   2048
#define BBATCH 2
#define SEQB   (BBATCH * LSEQ)   // 4096 rows total

#define KTILE   64               // K-depth staged per LDS tile
#define LDSS    72               // padded LDS row stride in bf16 elements (144B)

// ---------------------------------------------------------------------------
// Scalar helpers
// ---------------------------------------------------------------------------
__device__ __forceinline__ bf16_t f2bf(float f) {
  union { float f; uint32_t u; } x; x.f = f;
  uint32_t r = x.u + 0x7FFFu + ((x.u >> 16) & 1u);   // round-to-nearest-even
  uint16_t h = (uint16_t)(r >> 16);
  bf16_t out; __builtin_memcpy(&out, &h, sizeof(out));
  return out;
}
__device__ __forceinline__ float bf2f(bf16_t h) {
  uint16_t u; __builtin_memcpy(&u, &h, sizeof(u));
  union { uint32_t u; float f; } x; x.u = ((uint32_t)u) << 16;
  return x.f;
}
__device__ __forceinline__ float silu_f(float x) {
  return x / (1.0f + __expf(-x));
}

// ---------------------------------------------------------------------------
// Tensor Data Mover (ISA ch.8 / 15): one descriptor-driven DMA per tile.
// Loads an nrows x 64 bf16 tile (row-major, element stride ldg) into LDS with
// native padding: row = 64 elem = 32 DWORDs -> pad_interval=4 (32 DW),
// pad_amount=3 (4 DW = 16B)  =>  LDS row stride 144B == LDSS elements.
// Issue from ONE wave only; completion via TENSORcnt.
// ---------------------------------------------------------------------------
__device__ __forceinline__ void tdm_load_tile(const bf16_t* g, int ldg,
                                              int row0, int k0,
                                              bf16_t* lds, int nrows) {
  uint64_t ga = (uint64_t)(g + (size_t)row0 * ldg + k0);
  uint32_t la = (uint32_t)(uintptr_t)lds;          // LDS byte offset
  u32x4 g0;
  g0[0] = 1u;                                      // count=1, user descriptor
  g0[1] = la;                                      // lds_addr
  g0[2] = (uint32_t)ga;                            // global_addr[31:0]
  g0[3] = (uint32_t)((ga >> 32) & 0x01FFFFFFu) | (2u << 30); // ga[56:32]|type=2
  u32x8 g1;
  uint32_t td0 = (uint32_t)ldg;                    // tensor_dim0 (elements)
  uint32_t td1 = 0x7FFFFFFFu;                      // tensor_dim1: no OOB clip
  uint64_t st0 = (uint64_t)ldg;                    // tensor_dim0_stride
  // w0: wg_mask=0 | data_size=1(2B)<<16 | pad_enable<<20 | pad_int=4<<22 | pad_amt=3<<25
  g1[0] = (1u << 16) | (1u << 20) | (4u << 22) | (3u << 25);
  g1[1] = (td0 & 0xFFFFu) << 16;                   // [31:16] = tensor_dim0[15:0]
  g1[2] = (td0 >> 16) | ((td1 & 0xFFFFu) << 16);   // td0 hi | td1 lo
  g1[3] = (td1 >> 16) | (64u << 16);               // td1 hi | tile_dim0=64
  g1[4] = (uint32_t)nrows;                         // tile_dim1 | tile_dim2=0
  g1[5] = (uint32_t)st0;                           // dim0_stride[31:0]
  g1[6] = (uint32_t)(st0 >> 32);                   // dim0_stride[47:32] | d1s lo=0
  g1[7] = 0u;                                      // tensor_dim1_stride hi
  asm volatile("tensor_load_to_lds %0, %1" :: "s"(g0), "s"(g1) : "memory");
}
__device__ __forceinline__ void wait_tensor0() {
  __builtin_amdgcn_s_wait_tensorcnt(0);
}

// ---------------------------------------------------------------------------
// CDNA5 per-lane async copy (ASYNCcnt) -- used by the attention kernel where
// each single-wave block stages its own 32x64 k/v tiles.
// ---------------------------------------------------------------------------
__device__ __forceinline__ void async_copy_tile32(const bf16_t* __restrict__ g,
                                                  int ldg, int row0,
                                                  bf16_t* lds) {
#pragma unroll
  for (int it = 0; it < 8; ++it) {                 // 256 chunks / 32 lanes
    int c = threadIdx.x + it * 32;
    int row = c >> 3, inner = c & 7;
    uint32_t laddr = (uint32_t)(uintptr_t)(lds + row * LDSS + inner * 8);
    uint64_t gaddr = (uint64_t)(g + (size_t)(row0 + row) * ldg + inner * 8);
    asm volatile("global_load_async_to_lds_b128 %0, %1, off"
                 :: "v"(laddr), "v"(gaddr) : "memory");
  }
}
__device__ __forceinline__ void wait_async0() {
  asm volatile("s_wait_asynccnt 0" ::: "memory");
}

// ---------------------------------------------------------------------------
// Fragment loaders (ISA 7.12.2 16-bit A layout): lane -> row (L&15),
// half (L>>4); e in [0,8): K = half*8+e ; e in [8,16): K = 16+half*8+(e-8).
// ---------------------------------------------------------------------------
__device__ __forceinline__ v16bf load_frag(const bf16_t* base,
                                           int row0, int k0, int ld) {
  int lane = threadIdx.x & 31;
  int r    = lane & 15;
  int half = lane >> 4;
  const bf16_t* p = base + (size_t)(row0 + r) * ld + k0 + half * 8;
  bf16x8 lo = *(const bf16x8*)(p);
  bf16x8 hi = *(const bf16x8*)(p + 16);
  v16bf f;
#pragma unroll
  for (int i = 0; i < 8; ++i) { f[i] = lo[i]; f[8 + i] = hi[i]; }
  return f;
}

// Gathered B fragment from a row-major [K_rows, ld] operand in LDS.
__device__ __forceinline__ v16bf load_frag_t(const bf16_t* base,
                                             int col0, int k0, int ld) {
  int lane = threadIdx.x & 31;
  int n    = lane & 15;
  int half = lane >> 4;
  v16bf f;
#pragma unroll
  for (int e = 0; e < 16; ++e) {
    int kk = (e < 8) ? (half * 8 + e) : (16 + half * 8 + (e - 8));
    f[e] = base[(size_t)(k0 + kk) * ld + col0 + n];
  }
  return f;
}

#define WMMA_BF16(a, b, c) \
  __builtin_amdgcn_wmma_f32_16x16x32_bf16(false, (a), false, (b), (short)0, (c), false, false)

// ---------------------------------------------------------------------------
// Weight convert + transpose: src fp32 [K, N] row-major -> dst bf16 [N, K]
// ---------------------------------------------------------------------------
__global__ void k_transpose_bf16(const float* __restrict__ src,
                                 bf16_t* __restrict__ dst, int K, int N) {
  long long idx   = (long long)blockIdx.x * blockDim.x + threadIdx.x;
  long long total = (long long)K * N;
  if (idx >= total) return;
  int k = (int)(idx % K);
  int n = (int)(idx / K);
  dst[(long long)n * K + k] = f2bf(src[(long long)k * N + n]);
}

// ---------------------------------------------------------------------------
// Row LayerNorm fp32 -> bf16 (one row of width D per block, 256 threads)
// ---------------------------------------------------------------------------
__global__ void k_layernorm_bf16(const float* __restrict__ x,
                                 const float* __restrict__ g,
                                 const float* __restrict__ b,
                                 bf16_t* __restrict__ out, int D) {
  int row = blockIdx.x;
  const float* xr = x + (long long)row * D;
  __shared__ float red[256];
  float s = 0.0f;
  for (int i = threadIdx.x; i < D; i += blockDim.x) s += xr[i];
  red[threadIdx.x] = s; __syncthreads();
  for (int st = 128; st > 0; st >>= 1) {
    if ((int)threadIdx.x < st) red[threadIdx.x] += red[threadIdx.x + st];
    __syncthreads();
  }
  float mu = red[0] / (float)D; __syncthreads();
  float v = 0.0f;
  for (int i = threadIdx.x; i < D; i += blockDim.x) { float d = xr[i] - mu; v += d * d; }
  red[threadIdx.x] = v; __syncthreads();
  for (int st = 128; st > 0; st >>= 1) {
    if ((int)threadIdx.x < st) red[threadIdx.x] += red[threadIdx.x + st];
    __syncthreads();
  }
  float rstd = rsqrtf(red[0] / (float)D + 1e-5f);
  for (int i = threadIdx.x; i < D; i += blockDim.x)
    out[(long long)row * D + i] = f2bf((xr[i] - mu) * rstd * g[i] + b[i]);
}

// ---------------------------------------------------------------------------
// TDM-staged, double-buffered GEMM: block = 8 waves, tile 128(M) x 64(N).
// A [M,K] bf16 row-major; BT [N,K] bf16 (pre-transposed weights).
// Wave 0 drives the Tensor Data Mover; fragments come from LDS.
// All fragment loads of a k-step are hoisted before the WMMAs so the ds waits
// amortize and the WMMAs can issue back-to-back.
// ---------------------------------------------------------------------------
__device__ __forceinline__ void gemm_step(const bf16_t* sA, const bf16_t* sB,
                                          int wrow, v8f acc[4]) {
#pragma unroll
  for (int kk = 0; kk < KTILE; kk += 32) {
    v16bf a  = load_frag(sA, wrow, kk, LDSS);
    v16bf b0 = load_frag(sB, 0,  kk, LDSS);
    v16bf b1 = load_frag(sB, 16, kk, LDSS);
    v16bf b2 = load_frag(sB, 32, kk, LDSS);
    v16bf b3 = load_frag(sB, 48, kk, LDSS);
    acc[0] = WMMA_BF16(a, b0, acc[0]);
    acc[1] = WMMA_BF16(a, b1, acc[1]);
    acc[2] = WMMA_BF16(a, b2, acc[2]);
    acc[3] = WMMA_BF16(a, b3, acc[3]);
  }
}

__device__ __forceinline__ void gemm_body(const bf16_t* __restrict__ A,
                                          const bf16_t* __restrict__ BT,
                                          int row0, int col0, int K,
                                          bf16_t sA[2][128 * LDSS],
                                          bf16_t sB[2][64 * LDSS],
                                          v8f acc[4]) {
  int wave = threadIdx.x >> 5;
  if (threadIdx.x < 32) {
    tdm_load_tile(A,  K, row0, 0, sA[0], 128);
    tdm_load_tile(BT, K, col0, 0, sB[0], 64);
    wait_tensor0();
  }
  __syncthreads();
  int nk = K / KTILE;
  int p = 0;
  for (int t = 0; t < nk; ++t) {
    if (t + 1 < nk && threadIdx.x < 32) {
      tdm_load_tile(A,  K, row0, (t + 1) * KTILE, sA[p ^ 1], 128);
      tdm_load_tile(BT, K, col0, (t + 1) * KTILE, sB[p ^ 1], 64);
    }
    gemm_step(sA[p], sB[p], wave * 16, acc);
    if (t + 1 < nk && threadIdx.x < 32) wait_tensor0();
    __syncthreads();
    p ^= 1;
  }
}

// C = silu(A @ BT^T), bf16 out. grid: (M/128, N/64), block 256 (8 waves)
__global__ void k_gemm_silu_bf16(const bf16_t* __restrict__ A,
                                 const bf16_t* __restrict__ BT,
                                 bf16_t* __restrict__ C, int M, int N, int K) {
  __shared__ __align__(16) bf16_t sA[2][128 * LDSS];
  __shared__ __align__(16) bf16_t sB[2][64 * LDSS];
  int wave = threadIdx.x >> 5;
  int row0 = blockIdx.x * 128;
  int col0 = blockIdx.y * 64;
  v8f acc[4] = {};
  gemm_body(A, BT, row0, col0, K, sA, sB, acc);
  int lane = threadIdx.x & 31, n = lane & 15, half = lane >> 4;
  int rw = row0 + wave * 16;
#pragma unroll
  for (int j = 0; j < 4; ++j)
#pragma unroll
    for (int i = 0; i < 8; ++i) {
      int r = rw + i + 8 * half, c = col0 + j * 16 + n;
      C[(long long)r * N + c] = f2bf(silu_f(acc[j][i]));
    }
}

// O = X + A @ BT^T, fp32 out/residual
__global__ void k_gemm_res_f32(const bf16_t* __restrict__ A,
                               const bf16_t* __restrict__ BT,
                               const float* __restrict__ X,
                               float* __restrict__ O, int M, int N, int K) {
  __shared__ __align__(16) bf16_t sA[2][128 * LDSS];
  __shared__ __align__(16) bf16_t sB[2][64 * LDSS];
  int wave = threadIdx.x >> 5;
  int row0 = blockIdx.x * 128;
  int col0 = blockIdx.y * 64;
  v8f acc[4] = {};
  gemm_body(A, BT, row0, col0, K, sA, sB, acc);
  int lane = threadIdx.x & 31, n = lane & 15, half = lane >> 4;
  int rw = row0 + wave * 16;
#pragma unroll
  for (int j = 0; j < 4; ++j)
#pragma unroll
    for (int i = 0; i < 8; ++i) {
      long long idx = (long long)(rw + i + 8 * half) * N + col0 + j * 16 + n;
      O[idx] = X[idx] + acc[j][i];
    }
}

// C = silu(A @ B1T^T) * (A @ B3T^T), bf16 out (SwiGLU hidden)
__global__ void k_gemm_gated_bf16(const bf16_t* __restrict__ A,
                                  const bf16_t* __restrict__ B1T,
                                  const bf16_t* __restrict__ B3T,
                                  bf16_t* __restrict__ C, int M, int N, int K) {
  __shared__ __align__(16) bf16_t sA[2][128 * LDSS];
  __shared__ __align__(16) bf16_t sB1[2][64 * LDSS];
  __shared__ __align__(16) bf16_t sB3[2][64 * LDSS];
  int wave = threadIdx.x >> 5;
  int row0 = blockIdx.x * 128;
  int col0 = blockIdx.y * 64;
  v8f acc1[4] = {}, acc3[4] = {};
  if (threadIdx.x < 32) {
    tdm_load_tile(A,   K, row0, 0, sA[0], 128);
    tdm_load_tile(B1T, K, col0, 0, sB1[0], 64);
    tdm_load_tile(B3T, K, col0, 0, sB3[0], 64);
    wait_tensor0();
  }
  __syncthreads();
  int nk = K / KTILE;
  int p = 0;
  for (int t = 0; t < nk; ++t) {
    if (t + 1 < nk && threadIdx.x < 32) {
      tdm_load_tile(A,   K, row0, (t + 1) * KTILE, sA[p ^ 1], 128);
      tdm_load_tile(B1T, K, col0, (t + 1) * KTILE, sB1[p ^ 1], 64);
      tdm_load_tile(B3T, K, col0, (t + 1) * KTILE, sB3[p ^ 1], 64);
    }
    gemm_step(sA[p], sB1[p], wave * 16, acc1);
    gemm_step(sA[p], sB3[p], wave * 16, acc3);
    if (t + 1 < nk && threadIdx.x < 32) wait_tensor0();
    __syncthreads();
    p ^= 1;
  }
  int lane = threadIdx.x & 31, n = lane & 15, half = lane >> 4;
  int rw = row0 + wave * 16;
#pragma unroll
  for (int j = 0; j < 4; ++j)
#pragma unroll
    for (int i = 0; i < 8; ++i) {
      int r = rw + i + 8 * half, c = col0 + j * 16 + n;
      C[(long long)r * N + c] = f2bf(silu_f(acc1[j][i]) * acc3[j][i]);
    }
}

// ---------------------------------------------------------------------------
// Attention: one wave per (b, h, 16-query tile); single-wave workgroup.
//   scores = silu((q k^T)/8) * causal ; a = scores @ v ; LN_d(a) * u
// k/v 32-key tiles async-staged to LDS (double buffered, ASYNCcnt).
// ---------------------------------------------------------------------------
__global__ void k_attention(const bf16_t* __restrict__ qkuv,
                            const float* __restrict__ nag,
                            const float* __restrict__ nab,
                            bf16_t* __restrict__ aout) {
  int qt = blockIdx.x;               // query tile (16 rows)
  int bh = blockIdx.y;
  int b = bh / NHEAD, h = bh % NHEAD;
  int q0 = qt * 16;

  const bf16_t* base = qkuv + (long long)b * LSEQ * FOURD + h * DHEAD;
  const bf16_t* qb = base;
  const bf16_t* kb = base + DMODEL;
  const bf16_t* vb = base + 2 * DMODEL;
  const bf16_t* ub = base + 3 * DMODEL;

  __shared__ __align__(16) bf16_t s_k[2][32 * LDSS];   // 32 keys x 64 dims
  __shared__ __align__(16) bf16_t s_v[2][32 * LDSS];
  __shared__ __align__(16) bf16_t s_sc[16 * 32];       // bf16 scores tile
  __shared__ float s_af[16 * DHEAD];                   // f32 output dump

  int lane = threadIdx.x & 31, n = lane & 15, half = lane >> 4;
  v8f acc[4] = {};
  const float scale = 0.125f;                          // 1/sqrt(64)
  int ktmax = (q0 + 15) >> 5;                          // causal bound

  v16bf aq0 = load_frag(qb, q0, 0,  FOURD);            // loop-invariant q
  v16bf aq1 = load_frag(qb, q0, 32, FOURD);

  async_copy_tile32(kb, FOURD, 0, s_k[0]);
  async_copy_tile32(vb, FOURD, 0, s_v[0]);
  wait_async0();
  int p = 0;

  for (int kt = 0; kt <= ktmax; ++kt) {
    if (kt + 1 <= ktmax) {
      async_copy_tile32(kb, FOURD, (kt + 1) * 32, s_k[p ^ 1]);
      async_copy_tile32(vb, FOURD, (kt + 1) * 32, s_v[p ^ 1]);
    }
    int ks = kt * 32;
    // ---- scores tile 16x32 = q[16 x 64] @ k^T[64 x 32] ----
    v8f sacc[2] = {};
    {
      v16bf bk00 = load_frag(s_k[p], 0,  0,  LDSS);
      v16bf bk10 = load_frag(s_k[p], 16, 0,  LDSS);
      v16bf bk01 = load_frag(s_k[p], 0,  32, LDSS);
      v16bf bk11 = load_frag(s_k[p], 16, 32, LDSS);
      sacc[0] = WMMA_BF16(aq0, bk00, sacc[0]);
      sacc[1] = WMMA_BF16(aq0, bk10, sacc[1]);
      sacc[0] = WMMA_BF16(aq1, bk01, sacc[0]);
      sacc[1] = WMMA_BF16(aq1, bk11, sacc[1]);
    }
    // ---- silu + causal mask -> LDS bf16 ----
#pragma unroll
    for (int j = 0; j < 2; ++j)
#pragma unroll
      for (int i = 0; i < 8; ++i) {
        int qi = q0 + i + 8 * half;
        int ki = ks + j * 16 + n;
        float sv  = sacc[j][i] * scale;
        float val = (ki <= qi) ? silu_f(sv) : 0.0f;
        s_sc[(i + 8 * half) * 32 + j * 16 + n] = f2bf(val);
      }
    // ---- a += scores[16 x 32] @ v[32 x 64] (v gathered from LDS) ----
    v16bf as = load_frag(s_sc, 0, 0, 32);
#pragma unroll
    for (int j = 0; j < 4; ++j) {
      v16bf bv = load_frag_t(s_v[p], j * 16, 0, LDSS);
      acc[j] = WMMA_BF16(as, bv, acc[j]);
    }
    if (kt + 1 <= ktmax) wait_async0();
    p ^= 1;
  }

  // ---- per-row LN over d=64, * u, store bf16 ----
#pragma unroll
  for (int j = 0; j < 4; ++j)
#pragma unroll
    for (int i = 0; i < 8; ++i)
      s_af[(i + 8 * half) * DHEAD + j * 16 + n] = acc[j][i];
  __syncthreads();

  if (lane < 16) {
    int row = lane, qrow = q0 + row;
    float mu = 0.0f;
    for (int j = 0; j < DHEAD; ++j) mu += s_af[row * DHEAD + j];
    mu *= (1.0f / DHEAD);
    float var = 0.0f;
    for (int j = 0; j < DHEAD; ++j) { float d = s_af[row * DHEAD + j] - mu; var += d * d; }
    var *= (1.0f / DHEAD);
    float rstd = rsqrtf(var + 1e-5f);
    for (int j = 0; j < DHEAD; ++j) {
      float uval = bf2f(ub[(long long)qrow * FOURD + j]);
      float av   = (s_af[row * DHEAD + j] - mu) * rstd * nag[j] + nab[j];
      aout[((long long)b * LSEQ + qrow) * DMODEL + h * DHEAD + j] = f2bf(av * uval);
    }
  }
}

// ---------------------------------------------------------------------------
// Host driver
// ---------------------------------------------------------------------------
extern "C" void kernel_launch(void* const* d_in, const int* in_sizes, int n_in,
                              void* d_out, int out_size, void* d_ws, size_t ws_size,
                              hipStream_t stream) {
  (void)in_sizes; (void)n_in; (void)out_size; (void)ws_size;
  const float* x      = (const float*)d_in[0];
  /* d_in[1] attn_mask: causal mask computed analytically */
  const float* w_qkuv = (const float*)d_in[2];
  const float* w_out  = (const float*)d_in[3];
  const float* w1     = (const float*)d_in[4];
  const float* w2     = (const float*)d_in[5];
  const float* w3     = (const float*)d_in[6];
  const float* ln1_g  = (const float*)d_in[7];
  const float* ln1_b  = (const float*)d_in[8];
  const float* ln2_g  = (const float*)d_in[9];
  const float* ln2_b  = (const float*)d_in[10];
  const float* na_g   = (const float*)d_in[11];
  const float* na_b   = (const float*)d_in[12];
  float* out = (float*)d_out;

  char* ws = (char*)d_ws;
  size_t off = 0;
  auto alloc = [&](size_t bytes) -> void* {
    void* p = ws + off;
    off += (bytes + 255) & ~(size_t)255;
    return p;
  };
  bf16_t* wqkuvT = (bf16_t*)alloc((size_t)FOURD * DMODEL * 2);
  bf16_t* woutT  = (bf16_t*)alloc((size_t)DMODEL * DMODEL * 2);
  bf16_t* w1T    = (bf16_t*)alloc((size_t)FDIM * DMODEL * 2);
  bf16_t* w3T    = (bf16_t*)alloc((size_t)FDIM * DMODEL * 2);
  bf16_t* w2T    = (bf16_t*)alloc((size_t)DMODEL * FDIM * 2);
  bf16_t* xn     = (bf16_t*)alloc((size_t)SEQB * DMODEL * 2);
  bf16_t* qkuv   = (bf16_t*)alloc((size_t)SEQB * FOURD * 2);
  bf16_t* abuf   = (bf16_t*)alloc((size_t)SEQB * DMODEL * 2);
  float*  x1     = (float*) alloc((size_t)SEQB * DMODEL * 4);
  bf16_t* xn2    = (bf16_t*)alloc((size_t)SEQB * DMODEL * 2);
  bf16_t* hbuf   = (bf16_t*)alloc((size_t)SEQB * FDIM * 2);

  const int TB = 256;
  auto blocks = [](long long total, int tb) { return (unsigned)((total + tb - 1) / tb); };

  // Weight downconvert + transpose (src [K,N] fp32 -> dst [N,K] bf16)
  k_transpose_bf16<<<blocks((long long)DMODEL * FOURD, TB), TB, 0, stream>>>(w_qkuv, wqkuvT, DMODEL, FOURD);
  k_transpose_bf16<<<blocks((long long)DMODEL * DMODEL, TB), TB, 0, stream>>>(w_out,  woutT,  DMODEL, DMODEL);
  k_transpose_bf16<<<blocks((long long)DMODEL * FDIM,   TB), TB, 0, stream>>>(w1,     w1T,    DMODEL, FDIM);
  k_transpose_bf16<<<blocks((long long)DMODEL * FDIM,   TB), TB, 0, stream>>>(w3,     w3T,    DMODEL, FDIM);
  k_transpose_bf16<<<blocks((long long)FDIM * DMODEL,   TB), TB, 0, stream>>>(w2,     w2T,    FDIM,   DMODEL);

  // LN1 -> xn (bf16)
  k_layernorm_bf16<<<SEQB, TB, 0, stream>>>(x, ln1_g, ln1_b, xn, DMODEL);

  // qkuv = silu(xn @ w_qkuv)
  {
    dim3 g(SEQB / 128, FOURD / 64);
    k_gemm_silu_bf16<<<g, TB, 0, stream>>>(xn, wqkuvT, qkuv, SEQB, FOURD, DMODEL);
  }

  // attention + head-LN * u -> abuf (bf16 [B,L,D])
  {
    dim3 g(LSEQ / 16, BBATCH * NHEAD);
    k_attention<<<g, 32, 0, stream>>>(qkuv, na_g, na_b, abuf);
  }

  // x1 = x + abuf @ w_out
  {
    dim3 g(SEQB / 128, DMODEL / 64);
    k_gemm_res_f32<<<g, TB, 0, stream>>>(abuf, woutT, x, x1, SEQB, DMODEL, DMODEL);
  }

  // LN2 -> xn2 (bf16)
  k_layernorm_bf16<<<SEQB, TB, 0, stream>>>(x1, ln2_g, ln2_b, xn2, DMODEL);

  // hbuf = silu(xn2 @ w1) * (xn2 @ w3)
  {
    dim3 g(SEQB / 128, FDIM / 64);
    k_gemm_gated_bf16<<<g, TB, 0, stream>>>(xn2, w1T, w3T, hbuf, SEQB, FDIM, DMODEL);
  }

  // out = x1 + hbuf @ w2
  {
    dim3 g(SEQB / 128, DMODEL / 64);
    k_gemm_res_f32<<<g, TB, 0, stream>>>(hbuf, w2T, x1, out, SEQB, DMODEL, FDIM);
  }
}
